// Attention_3109556322333
// MI455X (gfx1250) — matrix operand order repaired and checked by reading, not compiled, not run on hardware
//
#include <hip/hip_runtime.h>
#include <stdint.h>

typedef __attribute__((ext_vector_type(16))) _Float16 v16h;
typedef __attribute__((ext_vector_type(8)))  float    v8f;
typedef __attribute__((ext_vector_type(4)))  unsigned int v4u;
typedef __attribute__((ext_vector_type(8)))  int      v8i_t;
typedef __attribute__((ext_vector_type(4)))  int      v4i_t;

#define B_    2
#define H_    4
#define D_    32
#define S_    4096
#define CIN   64
#define HID   128
#define EPS_   1e-5f
// SCALE * log2(e): softmax is computed in base-2 (v_exp_f32 is natively 2^x)
#define QSCALE_ 23.083120654223414f

union F16x16 { v16h v; uint32_t u[8]; _Float16 h[16]; };
union U32H  { uint32_t u; _Float16 h[2]; };

// ---------------------------------------------------------------------------
// TDM: DMA a 2D tile (tile_d0 x tile_d1 elements of 2 bytes) from global to
// LDS, written linearly. D# per CDNA5 ISA.
// ---------------------------------------------------------------------------
__device__ __forceinline__ void tdm_load_2d(uint32_t lds_addr, const void* gptr,
    uint32_t tile_d0, uint32_t tile_d1,
    uint32_t tensor_d0, uint32_t tensor_d1, uint32_t stride0)
{
  uint64_t ga = (uint64_t)(uintptr_t)gptr;
  v4u g0;
  g0.x = 1u;                                   // count=1, user descriptor
  g0.y = lds_addr;                             // lds_addr (bytes)
  g0.z = (uint32_t)ga;                         // global_addr[31:0]
  g0.w = (uint32_t)(ga >> 32) | (2u << 30);    // global_addr[56:32] | type=2
  v8i_t g1;
  g1[0] = (int)(1u << 16);                     // workgroup_mask=0, data_size=1 (2B)
  g1[1] = (int)(tensor_d0 << 16);              // abar=0 | tensor_dim0[15:0]
  g1[2] = (int)((tensor_d0 >> 16) | (tensor_d1 << 16));
  g1[3] = (int)((tensor_d1 >> 16) | (tile_d0 << 16));
  g1[4] = (int)tile_d1;                        // tile_dim1 | tile_dim2=0
  g1[5] = (int)stride0;                        // tensor_dim0_stride[31:0]
  g1[6] = 0;
  g1[7] = 0;
  v4i_t z4 = {0, 0, 0, 0};
#if defined(__clang_major__) && __clang_major__ >= 23
  v8i_t z8 = {0, 0, 0, 0, 0, 0, 0, 0};
  __builtin_amdgcn_tensor_load_to_lds(g0, g1, z4, z4, z8, 0);
#else
  __builtin_amdgcn_tensor_load_to_lds(g0, g1, z4, z4, 0);
#endif
}

// ---------------------------------------------------------------------------
// Kernel 1: qkv = input @ w_qkv ; q,k -> f32 [bh][s][32], v -> f16 TRANSPOSED
// [bh][d][s] so the V tile is a plain 2D TDM tile landing as vt[ch][key].
// ---------------------------------------------------------------------------
__global__ __launch_bounds__(128) void qkv_kernel(
    const float* __restrict__ inp, const float* __restrict__ wqkv,
    float* __restrict__ qf, float* __restrict__ kf, _Float16* __restrict__ vhT)
{
  int bs = blockIdx.x;           // b*S + s
  int t  = threadIdx.x;          // 0..127
  __shared__ float inrow[CIN];
  if (t < CIN) inrow[t] = inp[(size_t)bs * CIN + t];
  __syncthreads();
  float a0 = 0.f, a1 = 0.f, a2 = 0.f;
#pragma unroll 8
  for (int c = 0; c < CIN; c++) {
    float ic = inrow[c];
    const float* w = wqkv + (size_t)c * 384;
    a0 += ic * w[t];
    a1 += ic * w[t + 128];
    a2 += ic * w[t + 256];
  }
  int b = bs >> 12, s = bs & (S_ - 1);
  int h = t >> 5, d = t & 31;
  int bh = b * H_ + h;
  size_t idx = (((size_t)bh) * S_ + s) * D_ + d;
  qf[idx] = a0;
  kf[idx] = a1;
  vhT[((size_t)bh * D_ + d) * S_ + s] = (_Float16)a2;
}

// ---------------------------------------------------------------------------
// Kernel 2: per-channel mean / rsqrt(var+eps) over s=4096 (512 channels)
// ---------------------------------------------------------------------------
__global__ __launch_bounds__(256) void stats_kernel(
    const float* __restrict__ qf, const float* __restrict__ kf,
    float* __restrict__ mu, float* __restrict__ rs)
{
  int idx = blockIdx.x;                       // 0..511
  const float* src = (idx < 256) ? qf : kf;
  int ch = idx & 255;                         // bh*32 + d
  const float* base = src + (size_t)(ch >> 5) * S_ * D_ + (ch & 31);
  float s1 = 0.f, s2 = 0.f;
  for (int i = threadIdx.x; i < S_; i += 256) {
    float v = base[(size_t)i * D_];
    s1 += v; s2 += v * v;
  }
  __shared__ float r1[256], r2[256];
  int t = threadIdx.x;
  r1[t] = s1; r2[t] = s2;
  __syncthreads();
  for (int off = 128; off > 0; off >>= 1) {
    if (t < off) { r1[t] += r1[t + off]; r2[t] += r2[t + off]; }
    __syncthreads();
  }
  if (t == 0) {
    float m   = r1[0] * (1.0f / S_);
    float var = r2[0] * (1.0f / S_) - m * m;
    mu[idx] = m;
    rs[idx] = rsqrtf(var + EPS_);
  }
}

// ---------------------------------------------------------------------------
// Kernel 3: standardize, convert to f16; q gets SCALE*log2(e) folded in
// ---------------------------------------------------------------------------
__global__ __launch_bounds__(256) void std_kernel(
    const float* __restrict__ qf, const float* __restrict__ kf,
    const float* __restrict__ mu, const float* __restrict__ rs,
    _Float16* __restrict__ qh, _Float16* __restrict__ kh)
{
  int i   = blockIdx.x * 256 + threadIdx.x;   // 0 .. 2*2^20-1
  int sel = i >> 20;                          // 0 = q, 1 = k
  int j   = i & ((1 << 20) - 1);              // [bh][s][d] flat
  int d   = j & 31;
  int bh  = j >> 17;
  int chIdx = sel * 256 + bh * 32 + d;
  float v = (sel ? kf : qf)[j];
  float o = (v - mu[chIdx]) * rs[chIdx] * (sel ? 1.0f : QSCALE_);
  (sel ? kh : qh)[j] = (_Float16)o;
}

// ---------------------------------------------------------------------------
// Kernel 4: flash attention, TRANSPOSED formulation. S^T = K Q^T so each lane
// owns 32 of the 64 scores of one query (partner lane^16 owns the rest):
// row max is in-lane + ONE shfl_xor(16); P^T B-fragments are assembled with
// 4 shfl_xor(16) per key group (no LDS roundtrip); O^T = V^T P^T and the
// softmax denominator rides a ones-matrix WMMA. K/V tiles are TDM
// double-buffered (tensor_load_to_lds + s_wait_tensorcnt).
// ---------------------------------------------------------------------------
__global__ __launch_bounds__(256) void flash_kernel(
    const _Float16* __restrict__ qh, const _Float16* __restrict__ kh,
    const _Float16* __restrict__ vhT, float* __restrict__ Og)
{
  __shared__ _Float16 kt[2][64 * 32];     // [key][ch] tiles
  __shared__ _Float16 vt[2][32 * 64];     // [ch][key] tiles (global V is [d][s])

  int tid  = threadIdx.x;
  int lane = tid & 31, wave = tid >> 5;
  int n = lane & 15, hs = lane >> 4;
  int bh = blockIdx.y;
  int b  = bh >> 2, h = bh & 3;
  int qbase = (blockIdx.x * 8 + wave) * 16;

  const _Float16* kbase = kh  + (size_t)bh * S_ * D_;   // [s][32]
  const _Float16* vbase = vhT + (size_t)bh * D_ * S_;   // [32][s]
  const uint32_t* q32 = (const uint32_t*)qh;

  // ---- Q as B-fragment (32ch x 16q): lane n = query, K = hs*16 + e
  F16x16 qb;
  {
    size_t rowu = ((size_t)bh * S_ + qbase + n) * 16;   // row in u32 units
#pragma unroll
    for (int j = 0; j < 8; j++)
      qb.u[j] = q32[rowu + hs * 8 + j];
  }

  // all-ones A fragment: softmax denominator via the matrix pipe
  F16x16 onef;
#pragma unroll
  for (int e = 0; e < 16; e++) onef.h[e] = (_Float16)1.0f;

  v8f o0 = {}, o1 = {}, o2 = {};   // O^T ch 0-15, ch 16-31, denominator
  float mrow = -1e30f;             // per-lane scalar: one query per lane

  // ---- prologue: TDM-stage tile 0 (wave 0 only; TENSORcnt is per-wave)
  if (wave == 0) {
    tdm_load_2d((uint32_t)(uintptr_t)&kt[0][0], kbase, 32, 64, 32, S_, 32);
    tdm_load_2d((uint32_t)(uintptr_t)&vt[0][0], vbase, 64, 32, S_, 32, S_);
    __builtin_amdgcn_s_wait_tensorcnt(0);
  }
  __syncthreads();

  const int NT = S_ / 64;
  for (int kb = 0; kb < NT; kb++) {
    int cur = kb & 1;
    if (wave == 0 && kb + 1 < NT) {
      int keynext = (kb + 1) * 64;
      tdm_load_2d((uint32_t)(uintptr_t)&kt[cur ^ 1][0],
                  kbase + (size_t)keynext * D_, 32, 64, 32, S_, 32);
      tdm_load_2d((uint32_t)(uintptr_t)&vt[cur ^ 1][0],
                  vbase + keynext,              64, 32, S_, 32, S_);
    }
    uint32_t* kt32 = (uint32_t*)&kt[cur][0];
    uint32_t* vt32 = (uint32_t*)&vt[cur][0];

    // ---- S^T chunks: s[c] = K[16 keys x 32 ch] (A) x Q^T (B)
    v8f s[4];
#pragma unroll
    for (int c = 0; c < 4; c++) {
      F16x16 ka;   // A frag: lane m = key-in-chunk; kt row is contiguous
#pragma unroll
      for (int j = 0; j < 8; j++)
        ka.u[j] = kt32[(c * 16 + n) * 16 + hs * 8 + j];
      v8f z = {};
      s[c] = __builtin_amdgcn_wmma_f32_16x16x32_f16(false, ka.v, false, qb.v, (short)0, z, false, false);
    }

    // ---- online softmax (base-2): in-lane max + one cross-half exchange
    float mx = -1e30f;
#pragma unroll
    for (int c = 0; c < 4; c++)
#pragma unroll
      for (int r = 0; r < 8; r++) mx = fmaxf(mx, s[c][r]);
    mx = fmaxf(mx, __shfl_xor(mx, 16, 32));
    float mnew = fmaxf(mrow, mx);
    float corr = exp2f(mrow - mnew);
    mrow = mnew;
#pragma unroll
    for (int r = 0; r < 8; r++) { o0[r] *= corr; o1[r] *= corr; o2[r] *= corr; }

    // ---- exp + pack to f16 pairs (chunk c, u32 j holds keys c*16+8*hs+2j..+1)
    uint32_t pk[4][4];
#pragma unroll
    for (int c = 0; c < 4; c++)
#pragma unroll
      for (int j = 0; j < 4; j++) {
        U32H t;
        t.h[0] = (_Float16)exp2f(s[c][2 * j]     - mnew);
        t.h[1] = (_Float16)exp2f(s[c][2 * j + 1] - mnew);
        pk[c][j] = t.u;
      }

    // ---- assemble P^T B-fragments per 32-key group with 4 shfl_xor(16)
    F16x16 pb[2];
#pragma unroll
    for (int g = 0; g < 2; g++)
#pragma unroll
      for (int j = 0; j < 4; j++) {
        uint32_t keep = hs ? pk[2 * g + 1][j] : pk[2 * g][j];
        uint32_t send = hs ? pk[2 * g][j]     : pk[2 * g + 1][j];
        uint32_t recv = (uint32_t)__shfl_xor((int)send, 16, 32);
        pb[g].u[j]     = hs ? recv : keep;
        pb[g].u[j + 4] = hs ? keep : recv;
      }

    // ---- V^T A-fragments (rows = channels) from the [ch][key] tile
    F16x16 va00, va01, va10, va11;  // (ch group, key group)
#pragma unroll
    for (int j = 0; j < 8; j++) {
      int e = hs * 8 + j;
      va00.u[j] = vt32[n        * 32 + e];        // ch 0-15,  keys 0-31
      va01.u[j] = vt32[n        * 32 + 16 + e];   // ch 0-15,  keys 32-63
      va10.u[j] = vt32[(16 + n) * 32 + e];        // ch 16-31, keys 0-31
      va11.u[j] = vt32[(16 + n) * 32 + 16 + e];
    }
    o0 = __builtin_amdgcn_wmma_f32_16x16x32_f16(false, va00.v, false, pb[0].v, (short)0, o0, false, false);
    o1 = __builtin_amdgcn_wmma_f32_16x16x32_f16(false, va10.v, false, pb[0].v, (short)0, o1, false, false);
    o2 = __builtin_amdgcn_wmma_f32_16x16x32_f16(false, onef.v, false, pb[0].v, (short)0, o2, false, false);
    o0 = __builtin_amdgcn_wmma_f32_16x16x32_f16(false, va01.v, false, pb[1].v, (short)0, o0, false, false);
    o1 = __builtin_amdgcn_wmma_f32_16x16x32_f16(false, va11.v, false, pb[1].v, (short)0, o1, false, false);
    o2 = __builtin_amdgcn_wmma_f32_16x16x32_f16(false, onef.v, false, pb[1].v, (short)0, o2, false, false);

    // TDM for the next tile must land before anyone flips buffers
    if (wave == 0) __builtin_amdgcn_s_wait_tensorcnt(0);
    __syncthreads();
  }

  // ---- epilogue: O^T[ch][q] / l; lane n = query, element r -> ch r+8*hs
  float inv = 1.0f / o2[0];
  int row = qbase + n;
  size_t oidx = ((size_t)b * S_ + row) * HID + h * D_;
#pragma unroll
  for (int r = 0; r < 8; r++) {
    Og[oidx + r + 8 * hs]      = o0[r] * inv;
    Og[oidx + 16 + r + 8 * hs] = o1[r] * inv;
  }
}

// ---------------------------------------------------------------------------
// Kernel 5: out = O @ w_out + b_out  (128 -> 64)
// ---------------------------------------------------------------------------
__global__ __launch_bounds__(64) void proj_kernel(
    const float* __restrict__ Og, const float* __restrict__ wout,
    const float* __restrict__ bout, float* __restrict__ out)
{
  int bs = blockIdx.x;
  int t  = threadIdx.x;  // 0..63
  __shared__ float orow[HID];
  orow[t]      = Og[(size_t)bs * HID + t];
  orow[t + 64] = Og[(size_t)bs * HID + t + 64];
  __syncthreads();
  float acc = bout[t];
#pragma unroll 8
  for (int c = 0; c < HID; c++) acc += orow[c] * wout[c * 64 + t];
  out[(size_t)bs * 64 + t] = acc;
}

// ---------------------------------------------------------------------------
extern "C" void kernel_launch(void* const* d_in, const int* in_sizes, int n_in,
                              void* d_out, int out_size, void* d_ws, size_t ws_size,
                              hipStream_t stream)
{
  (void)in_sizes; (void)n_in; (void)out_size; (void)ws_size;
  const float* inp  = (const float*)d_in[0];   // [2,16,16,16,64]
  const float* wqkv = (const float*)d_in[1];   // [64,384]
  const float* wout = (const float*)d_in[2];   // [128,64]
  const float* bout = (const float*)d_in[3];   // [64]
  float* out = (float*)d_out;                  // [2,16,16,16,64]

  char* ws = (char*)d_ws;
  size_t off = 0;
  const size_t QKV_F32 = (size_t)B_ * H_ * S_ * D_ * sizeof(float);     // 4 MB
  const size_t QKV_F16 = (size_t)B_ * H_ * S_ * D_ * sizeof(_Float16);  // 2 MB
  float*    qf  = (float*)(ws + off);    off += QKV_F32;
  float*    kf  = (float*)(ws + off);    off += QKV_F32;
  float*    mu  = (float*)(ws + off);    off += 4096;   // 512 floats (padded)
  float*    rs  = (float*)(ws + off);    off += 4096;
  _Float16* qh  = (_Float16*)(ws + off); off += QKV_F16;
  _Float16* kh  = (_Float16*)(ws + off); off += QKV_F16;
  _Float16* vhT = (_Float16*)(ws + off); off += QKV_F16;
  float*    Og  = (float*)(ws + off);    off += (size_t)B_ * S_ * HID * sizeof(float);

  qkv_kernel  <<<B_ * S_, 128, 0, stream>>>(inp, wqkv, qf, kf, vhT);
  stats_kernel<<<512,     256, 0, stream>>>(qf, kf, mu, rs);
  std_kernel  <<<(2 * B_ * H_ * S_ * D_) / 256, 256, 0, stream>>>(qf, kf, mu, rs, qh, kh);
  flash_kernel<<<dim3(S_ / 16 / 8, B_ * H_), 256, 0, stream>>>(qh, kh, vhT, Og);
  proj_kernel <<<B_ * S_, 64, 0, stream>>>(Og, wout, bout, out);
}
